// PhysicalSelfAttention_23776938951102
// MI455X (gfx1250) — matrix operand order repaired and checked by reading, not compiled, not run on hardware
//
#include <hip/hip_runtime.h>

typedef __bf16 bf16_t;
typedef __attribute__((ext_vector_type(16))) __bf16 v16bf;
typedef __attribute__((ext_vector_type(8)))  __bf16 v8bf;
typedef __attribute__((ext_vector_type(8)))  float  v8f;

#define WMMA_BF16(A, B, Cacc) \
  __builtin_amdgcn_wmma_f32_16x16x32_bf16(false, (A), false, (B), (short)0, (Cacc), false, false)

// Problem sizes
constexpr int Bn = 8, CIN = 512, Hin = 66, Win = 66;
constexpr int C = 256, CBAR = 32;
constexpr int HO = 64, WO = 64, N = HO * WO;      // 4096
constexpr int K9 = CIN * 9;                       // 4608
constexpr int KT_CONV = K9 / 32;                  // 144

// Combine two contiguous 8-element halves into a v16bf A/B fragment register.
__device__ __forceinline__ v16bf join16(v8bf lo, v8bf hi) {
  v16bf r;
#pragma unroll
  for (int i = 0; i < 8; ++i) { r[i] = lo[i]; r[i + 8] = hi[i]; }
  return r;
}

// ---------------------------------------------------------------------------
// f32 -> bf16 elementwise convert
__global__ void cvt_bf16_kernel(const float* __restrict__ in, bf16_t* __restrict__ out, int n) {
  int i = blockIdx.x * blockDim.x + threadIdx.x;
  if (i < n) out[i] = (bf16_t)in[i];
}

// ---------------------------------------------------------------------------
// Pack W_fcn [256][4608] into per-lane A-fragment order (16x32 bf16 tiles).
// A fragment (16-bit, 16x32): lane m = lane&15, half = lane>=16,
//   element i -> K = i + 16*half (contiguous order used here, not the ISA table order)
__global__ void pack_wfcn_kernel(const float* __restrict__ w, bf16_t* __restrict__ wp) {
  int lane = threadIdx.x, half = lane >> 4, m = lane & 15;
  int cot = blockIdx.x;   // 0..15
  int kt  = blockIdx.y;   // 0..143
  int co = cot * 16 + m;
  bf16_t* dst = wp + (((size_t)cot * KT_CONV + kt) * 32 + lane) * 16;
#pragma unroll
  for (int i = 0; i < 16; ++i) {
    int k = kt * 32 + i + 16 * half;
    dst[i] = (bf16_t)w[(size_t)co * K9 + k];
  }
}

// Pack Wf/Wg/Wh [32][256] -> wp[proj][t(2)][kt(8)][lane][16]
__global__ void pack_wproj_kernel(const float* __restrict__ Wf, const float* __restrict__ Wg,
                                  const float* __restrict__ Wh, bf16_t* __restrict__ wp) {
  int lane = threadIdx.x, half = lane >> 4, m = lane & 15;
  int proj = blockIdx.x, t = blockIdx.y, kt = blockIdx.z;
  const float* W = (proj == 0) ? Wf : ((proj == 1) ? Wg : Wh);
  int row = t * 16 + m;
  bf16_t* dst = wp + ((((size_t)(proj * 2 + t) * 8 + kt) * 32) + lane) * 16;
#pragma unroll
  for (int i = 0; i < 16; ++i) {
    int k = kt * 32 + i + 16 * half;
    dst[i] = (bf16_t)W[row * C + k];
  }
}

// Pack Wv [256][32] -> wp[rt(16)][lane][16]  (single K-tile, K=32)
__global__ void pack_wv_kernel(const float* __restrict__ Wv, bf16_t* __restrict__ wp) {
  int lane = threadIdx.x, half = lane >> 4, m = lane & 15;
  int rt = blockIdx.x;
  bf16_t* dst = wp + ((size_t)rt * 32 + lane) * 16;
#pragma unroll
  for (int i = 0; i < 16; ++i) {
    int k = i + 16 * half;  // < 32
    dst[i] = (bf16_t)Wv[(rt * 16 + m) * CBAR + k];
  }
}

// ---------------------------------------------------------------------------
// 3x3 VALID conv as implicit GEMM: out[co, p] = sum_k W[co,k] * im2col[k,p].
// One wave computes a 128-row (8 co-tiles) x 16-col strip so the expensive
// im2col gather of the B fragment is amortized over 8 WMMAs per K-step.
__global__ void conv3x3_wmma_kernel(const bf16_t* __restrict__ inb, const bf16_t* __restrict__ wp,
                                    const float* __restrict__ bias,
                                    float* __restrict__ xf, bf16_t* __restrict__ xb) {
  int lane = threadIdx.x, half = lane >> 4, col = lane & 15;
  int pt = blockIdx.x;    // 0..2047
  int cog = blockIdx.y;   // 0..1 (128 output channels each)
  int p = pt * 16 + col;
  int bb = p >> 12;
  int n  = p & (N - 1);
  int oh = n >> 6, ow = n & 63;
  const bf16_t* inbb = inb + (size_t)bb * CIN * Hin * Win;

  v8f acc[8];
#pragma unroll
  for (int t = 0; t < 8; ++t) acc[t] = (v8f){};

  for (int kt = 0; kt < KT_CONV; ++kt) {
    // B fragment (32x16, 16-bit): lane col = output position, K = i + 16*half
    v16bf bfrag;
    unsigned kbase = (unsigned)(kt * 32 + 16 * half);
#pragma unroll
    for (int i = 0; i < 16; ++i) {
      unsigned k  = kbase + i;
      unsigned ci = k / 9u;
      unsigned r  = k - ci * 9u;
      unsigned kh = r / 3u, kw = r - kh * 3u;
      bfrag[i] = inbb[(ci * Hin + (oh + kh)) * Win + (ow + kw)];
    }
#pragma unroll
    for (int t = 0; t < 8; ++t) {
      int cot = cog * 8 + t;
      v16bf a = *(const v16bf*)(wp + (((size_t)cot * KT_CONV + kt) * 32 + lane) * 16);
      acc[t] = WMMA_BF16(a, bfrag, acc[t]);
    }
  }
  // Epilogue: lane holds col = p; rows co = cog*128 + t*16 + r + 8*half.
#pragma unroll
  for (int t = 0; t < 8; ++t) {
    int co0 = cog * 128 + t * 16 + 8 * half;
    v8bf pk;
#pragma unroll
    for (int r = 0; r < 8; ++r) {
      float v = acc[t][r] + bias[co0 + r];
      v = v > 0.f ? v : 0.f;
      xf[((size_t)bb * C + co0 + r) * N + n] = v;          // f32 for residual
      pk[r] = (bf16_t)v;
    }
    // token-major bf16: 8 consecutive channels -> one 16B store
    *(v8bf*)&xb[((size_t)bb * N + n) * C + co0] = pk;
  }
}

// ---------------------------------------------------------------------------
// 1x1 projections: {f,g,h}x = W(32x256) @ x(256xN) + b.  One wave per
// (proj, batch, 16-token tile).  q (gx) and k (fx) stored token-major
// [b][n][32]; v (hx) stored channel-major [b][32][n] so attention PV
// B-fragments become contiguous 32B loads.
__global__ void proj_wmma_kernel(const bf16_t* __restrict__ xb, const bf16_t* __restrict__ wp,
                                 const float* __restrict__ bf_, const float* __restrict__ bg_,
                                 const float* __restrict__ bh_,
                                 bf16_t* __restrict__ qb,   // gx -> queries (token-major)
                                 bf16_t* __restrict__ kb,   // fx -> keys    (token-major)
                                 bf16_t* __restrict__ vt) { // hx -> values  (channel-major)
  int lane = threadIdx.x, half = lane >> 4, col = lane & 15;
  int nt = blockIdx.x, b = blockIdx.y, proj = blockIdx.z;
  const float* bias = (proj == 0) ? bf_ : ((proj == 1) ? bg_ : bh_);
  int n = nt * 16 + col;

  v8f acc0 = {}, acc1 = {};
  for (int kt = 0; kt < 8; ++kt) {
    v16bf bfrag = *(const v16bf*)(xb + ((size_t)b * N + n) * C + kt * 32 + 16 * half);
    v16bf a0 = *(const v16bf*)(wp + ((((size_t)(proj * 2 + 0) * 8 + kt) * 32) + lane) * 16);
    acc0 = WMMA_BF16(a0, bfrag, acc0);
    v16bf a1 = *(const v16bf*)(wp + ((((size_t)(proj * 2 + 1) * 8 + kt) * 32) + lane) * 16);
    acc1 = WMMA_BF16(a1, bfrag, acc1);
  }
  if (proj == 2) {
#pragma unroll
    for (int r = 0; r < 8; ++r) {
      int cb0 = r + 8 * half;
      vt[((size_t)b * CBAR + cb0) * N + n]      = (bf16_t)(acc0[r] + bias[cb0]);
      vt[((size_t)b * CBAR + 16 + cb0) * N + n] = (bf16_t)(acc1[r] + bias[16 + cb0]);
    }
  } else {
    bf16_t* out = (proj == 0) ? kb : qb;
#pragma unroll
    for (int r = 0; r < 8; ++r) {
      int cb0 = r + 8 * half;
      size_t base = ((size_t)b * N + n) * CBAR;
      out[base + cb0]      = (bf16_t)(acc0[r] + bias[cb0]);
      out[base + 16 + cb0] = (bf16_t)(acc1[r] + bias[16 + cb0]);
    }
  }
}

// ---------------------------------------------------------------------------
// Flash attention, head-dim 32.  s[m][n] = q_m . k_n, softmax over n,
// o[m][:] = sum_n p[m][n] v_n.  One wave per (batch, 16-query tile).
// Row stats use width-16 shuffles: wave32 C-fragment rows r / r+8 live in
// lane halves [0,16) / [16,32).
__global__ void attn_flash_kernel(const bf16_t* __restrict__ q, const bf16_t* __restrict__ kkey,
                                  const bf16_t* __restrict__ vt, bf16_t* __restrict__ o) {
  __shared__ bf16_t lds_p[16 * 32];
  int lane = threadIdx.x, half = lane >> 4, col = lane & 15;
  int mt = blockIdx.x, b = blockIdx.y;

  // Q as A-fragment: row m = mt*16 + (lane&15); K = i + 16*half.
  v16bf qa;
  {
    const bf16_t* qr = q + ((size_t)b * N + mt * 16 + col) * CBAR;
    qa = join16(*(const v8bf*)(qr + 16 * half), *(const v8bf*)(qr + 16 * half + 8));
  }

  float mrow[8], lrow[8];
  v8f o0 = {}, o1 = {};
#pragma unroll
  for (int r = 0; r < 8; ++r) { mrow[r] = -3.0e38f; lrow[r] = 0.f; }

  for (int n0 = 0; n0 < N; n0 += 32) {
    if (n0 + 32 < N) {  // prefetch next K/V tiles (global_prefetch path; data is L2-resident)
      __builtin_prefetch(kkey + ((size_t)b * N + n0 + 32 + col) * CBAR, 0, 1);
      __builtin_prefetch(vt + ((size_t)b * CBAR + col) * N + n0 + 32, 0, 1);
      __builtin_prefetch(vt + ((size_t)b * CBAR + 16 + col) * N + n0 + 32, 0, 1);
    }
    // Two 16-key score tiles (K = 32 channels each); contiguous 32B B loads
    v16bf k0 = *(const v16bf*)(kkey + ((size_t)b * N + n0 + col) * CBAR + 16 * half);
    v8f s0 = {};
    s0 = WMMA_BF16(qa, k0, s0);
    v16bf k1 = *(const v16bf*)(kkey + ((size_t)b * N + n0 + 16 + col) * CBAR + 16 * half);
    v8f s1 = {};
    s1 = WMMA_BF16(qa, k1, s1);

#pragma unroll
    for (int r = 0; r < 8; ++r) {
      float mx = fmaxf(s0[r], s1[r]);
#pragma unroll
      for (int msk = 1; msk < 16; msk <<= 1) mx = fmaxf(mx, __shfl_xor(mx, msk, 16));
      float mnew = fmaxf(mrow[r], mx);
      float corr = __expf(mrow[r] - mnew);
      float p0 = __expf(s0[r] - mnew);
      float p1 = __expf(s1[r] - mnew);
      float ps = p0 + p1;
#pragma unroll
      for (int msk = 1; msk < 16; msk <<= 1) ps += __shfl_xor(ps, msk, 16);
      lrow[r] = lrow[r] * corr + ps;
      mrow[r] = mnew;
      o0[r] *= corr;
      o1[r] *= corr;
      int mlocal = r + 8 * half;
      lds_p[mlocal * 32 + col]      = (bf16_t)p0;
      lds_p[mlocal * 32 + 16 + col] = (bf16_t)p1;
    }
    __syncthreads();

    // Re-layout P: C-fragment -> A-fragment via LDS (two ds b128 loads)
    v16bf pa = join16(*(const v8bf*)&lds_p[col * 32 + 16 * half],
                      *(const v8bf*)&lds_p[col * 32 + 16 * half + 8]);
    // V channel-major: B-fragment lane col = channel, K = consecutive tokens
    v16bf vb0 = *(const v16bf*)(vt + ((size_t)b * CBAR + col) * N + n0 + 16 * half);
    v16bf vb1 = *(const v16bf*)(vt + ((size_t)b * CBAR + 16 + col) * N + n0 + 16 * half);
    o0 = WMMA_BF16(pa, vb0, o0);
    o1 = WMMA_BF16(pa, vb1, o1);
    __syncthreads();
  }

#pragma unroll
  for (int r = 0; r < 8; ++r) {
    float inv = 1.f / lrow[r];
    int m = mt * 16 + r + 8 * half;
    size_t base = ((size_t)b * N + m) * CBAR;
    o[base + col]      = (bf16_t)(o0[r] * inv);
    o[base + 16 + col] = (bf16_t)(o1[r] * inv);
  }
}

// ---------------------------------------------------------------------------
// y = gamma * (Wv @ o + bv) + x.  One WMMA (K=32) per 16x16 tile, fused epilogue.
__global__ void out_wmma_kernel(const bf16_t* __restrict__ ob, const bf16_t* __restrict__ wvp,
                                const float* __restrict__ bv, const float* __restrict__ gamma,
                                const float* __restrict__ xf, float* __restrict__ y) {
  int lane = threadIdx.x, half = lane >> 4, col = lane & 15;
  int nt = blockIdx.x, rt = blockIdx.y, b = blockIdx.z;
  int n = nt * 16 + col;

  v16bf a = *(const v16bf*)(wvp + ((size_t)rt * 32 + lane) * 16);
  v16bf bfrag = *(const v16bf*)(ob + ((size_t)b * N + n) * CBAR + 16 * half);
  v8f acc = {};
  acc = WMMA_BF16(a, bfrag, acc);

  float g = gamma[0];
#pragma unroll
  for (int r = 0; r < 8; ++r) {
    int co = rt * 16 + r + 8 * half;
    size_t idx = ((size_t)b * C + co) * N + n;
    y[idx] = g * (acc[r] + bv[co]) + xf[idx];
  }
}

// ---------------------------------------------------------------------------
extern "C" void kernel_launch(void* const* d_in, const int* in_sizes, int n_in,
                              void* d_out, int out_size, void* d_ws, size_t ws_size,
                              hipStream_t stream) {
  (void)in_sizes; (void)n_in; (void)out_size; (void)ws_size;
  const float* feat  = (const float*)d_in[0];
  const float* Wfcn  = (const float*)d_in[1];
  const float* bfcn  = (const float*)d_in[2];
  const float* Wf    = (const float*)d_in[3];
  const float* bf_   = (const float*)d_in[4];
  const float* Wg    = (const float*)d_in[5];
  const float* bg_   = (const float*)d_in[6];
  const float* Wh    = (const float*)d_in[7];
  const float* bh_   = (const float*)d_in[8];
  const float* Wv    = (const float*)d_in[9];
  const float* bv_   = (const float*)d_in[10];
  const float* gamma = (const float*)d_in[11];
  float* y = (float*)d_out;

  char* ws = (char*)d_ws;
  size_t off = 0;
  auto suballoc = [&](size_t bytes) -> char* {
    char* p = ws + off;
    off = (off + bytes + 255) & ~(size_t)255;
    return p;
  };
  bf16_t* inb = (bf16_t*)suballoc((size_t)Bn * CIN * Hin * Win * 2);
  bf16_t* wfp = (bf16_t*)suballoc((size_t)16 * KT_CONV * 32 * 16 * 2);
  bf16_t* wpp = (bf16_t*)suballoc((size_t)3 * 2 * 8 * 32 * 16 * 2);
  bf16_t* wvp = (bf16_t*)suballoc((size_t)16 * 32 * 16 * 2);
  float*  xf  = (float*) suballoc((size_t)Bn * C * N * 4);
  bf16_t* xb  = (bf16_t*)suballoc((size_t)Bn * N * C * 2);
  bf16_t* qb  = (bf16_t*)suballoc((size_t)Bn * N * CBAR * 2);
  bf16_t* kb  = (bf16_t*)suballoc((size_t)Bn * N * CBAR * 2);
  bf16_t* vt  = (bf16_t*)suballoc((size_t)Bn * CBAR * N * 2);
  bf16_t* ob  = (bf16_t*)suballoc((size_t)Bn * N * CBAR * 2);

  int nin = Bn * CIN * Hin * Win;
  cvt_bf16_kernel<<<(nin + 255) / 256, 256, 0, stream>>>(feat, inb, nin);
  pack_wfcn_kernel<<<dim3(16, KT_CONV), 32, 0, stream>>>(Wfcn, wfp);
  pack_wproj_kernel<<<dim3(3, 2, 8), 32, 0, stream>>>(Wf, Wg, Wh, wpp);
  pack_wv_kernel<<<16, 32, 0, stream>>>(Wv, wvp);

  conv3x3_wmma_kernel<<<dim3((Bn * N) / 16, 2), 32, 0, stream>>>(inb, wfp, bfcn, xf, xb);
  proj_wmma_kernel<<<dim3(N / 16, Bn, 3), 32, 0, stream>>>(xb, wpp, bf_, bg_, bh_, qb, kb, vt);
  attn_flash_kernel<<<dim3(N / 16, Bn), 32, 0, stream>>>(qb, kb, vt, ob);
  out_wmma_kernel<<<dim3(N / 16, C / 16, Bn), 32, 0, stream>>>(ob, wvp, bv_, gamma, xf, y);
}
